// DecoderOnlyBlock_446676599027
// MI455X (gfx1250) — matrix operand order repaired and checked by reading, not compiled, not run on hardware
//
#include <hip/hip_runtime.h>
#include <math.h>
#include <stdint.h>

// ---------------------------------------------------------------------------
// Types for CDNA5 WMMA (gfx1250, wave32)
// ---------------------------------------------------------------------------
typedef __bf16 bf16;
typedef __attribute__((ext_vector_type(16))) __bf16 v16bf;
typedef __attribute__((ext_vector_type(8)))  float   v8f;
typedef __attribute__((ext_vector_type(8)))  __bf16 bf16x8;

static constexpr int kB = 2, kS = 2048, kE = 1024, kH = 16, kD = 64, kF = 4096;
static constexpr int kM = kB * kS;   // 4096 token rows

__device__ __forceinline__ v8f wmma_bf16(v16bf a, v16bf b, v8f c) {
  // D = A(16x32) * B(32x16) + C(16x16), f32 accumulate
  return __builtin_amdgcn_wmma_f32_16x16x32_bf16(false, a, false, b, (short)0, c,
                                                 false, false);
}

__device__ __forceinline__ v16bf concat8(bf16x8 lo, bf16x8 hi) {
  return __builtin_shufflevector(lo, hi, 0, 1, 2, 3, 4, 5, 6, 7,
                                 8, 9, 10, 11, 12, 13, 14, 15);
}

__device__ __forceinline__ float gelu_exact(float x) {
  return 0.5f * x * (1.0f + erff(x * 0.70710678118654752f));
}

// LDS byte offset of a generic pointer to __shared__ (aperture low 32 bits)
__device__ __forceinline__ uint32_t lds_off(const void* p) {
  return (uint32_t)(uintptr_t)p;
}

// Async global -> LDS copy of 16 bytes (CDNA5, tracked by ASYNCcnt).
// GVS addressing: mem = SGPR64 base + signed 32-bit VGPR offset.
__device__ __forceinline__ void async_copy_b128(uint32_t lds_byte_off,
                                                const void* base,
                                                int glb_byte_off) {
  asm volatile("global_load_async_to_lds_b128 %0, %1, %2"
               :: "v"(lds_byte_off), "v"(glb_byte_off), "s"(base)
               : "memory");
}

// ---------------------------------------------------------------------------
// fp32 [K][N] -> bf16 [N][K] transposed conversion (weights)
// ---------------------------------------------------------------------------
__global__ __launch_bounds__(256) void cvt_tr_bf16_kernel(const float* __restrict__ in,
                                                          bf16* __restrict__ out,
                                                          int Kdim, int Ndim) {
  size_t i = (size_t)blockIdx.x * 256 + threadIdx.x;
  if (i < (size_t)Kdim * Ndim) {
    int k = (int)(i / Ndim), n = (int)(i % Ndim);
    out[(size_t)n * Kdim + k] = (bf16)in[i];
  }
}

// ---------------------------------------------------------------------------
// LayerNorm over E (compile-time); one block per row. Optional fp32 + bf16 out.
// ---------------------------------------------------------------------------
template <int E>
__global__ __launch_bounds__(256) void ln_kernel(const float* __restrict__ x,
                                                 const float* __restrict__ g,
                                                 const float* __restrict__ bta,
                                                 float* __restrict__ outf,
                                                 bf16* __restrict__ outb) {
  __shared__ float red[256];
  const int tid = threadIdx.x;
  const float* xr = x + (size_t)blockIdx.x * E;
  float s = 0.f;
#pragma unroll
  for (int i = 0; i < E / 256; ++i) s += xr[tid + i * 256];
  red[tid] = s; __syncthreads();
  for (int o = 128; o > 0; o >>= 1) { if (tid < o) red[tid] += red[tid + o]; __syncthreads(); }
  const float mu = red[0] * (1.0f / E);
  __syncthreads();
  float v = 0.f;
#pragma unroll
  for (int i = 0; i < E / 256; ++i) { float d = xr[tid + i * 256] - mu; v += d * d; }
  red[tid] = v; __syncthreads();
  for (int o = 128; o > 0; o >>= 1) { if (tid < o) red[tid] += red[tid + o]; __syncthreads(); }
  const float rstd = rsqrtf(red[0] * (1.0f / E) + 1e-5f);
#pragma unroll
  for (int i = 0; i < E / 256; ++i) {
    const int col = tid + i * 256;
    float y = (xr[col] - mu) * rstd * g[col] + bta[col];
    size_t off = (size_t)blockIdx.x * E + col;
    if (outf) outf[off] = y;
    outb[off] = (bf16)y;
  }
}

// ---------------------------------------------------------------------------
// bf16 WMMA GEMM: C[M,N] = A[M,K] @ Wt[N,K]^T + bias, compile-time epilogue.
// A row-major [M,K]; weights pre-transposed [N,K] so BOTH operands are
// K-contiguous -> all fragment gathers are 16B vector LDS loads.
// Block tile 128x128, 8 waves (4x2), wave tile 32x64 (2x4 fragments), BK=32.
// Double-buffered async global->LDS staging (ASYNCcnt).
// EPI: 0 = bias -> bf16 [M,N]
//      1 = bias+GELU -> bf16 [M,N]
//      2 = bias+resid -> f32 [M,N]
//      3 = bias -> bf16 transposed [B,H,D,S]  (for attention V)
// ---------------------------------------------------------------------------
template <int EPI>
__global__ __launch_bounds__(256) void gemm_bf16_kernel(
    const bf16* __restrict__ A, const bf16* __restrict__ Wt,
    const float* __restrict__ bias, const float* __restrict__ resid,
    float* __restrict__ Cf, bf16* __restrict__ Cb,
    int Mdim, int Ndim, int Kdim) {
  constexpr int BM = 128, BN = 128, BK = 32;
  __shared__ __align__(16) bf16 As[2][BM][BK + 8];
  __shared__ __align__(16) bf16 Bs[2][BN][BK + 8];
  const int tid  = threadIdx.x;
  const int lane = tid & 31;
  const int w    = tid >> 5;
  const int wm   = w >> 1;        // 0..3 -> M offset wm*32
  const int wn   = w & 1;         // 0..1 -> N offset wn*64
  const int l16  = lane & 15;
  const int half = lane >> 4;
  const int mBase = blockIdx.y * BM;
  const int nBase = blockIdx.x * BN;
  v8f acc[2][4] = {};

  // Stage one 128x32 A tile + 128x32 Wt tile (4 async 16B copies per thread)
  auto stage = [&](int k0, int buf) {
#pragma unroll
    for (int it = 0; it < 2; ++it) {
      int idx = tid + it * 256;
      int r = idx >> 2, c = (idx & 3) * 8;
      async_copy_b128(lds_off(&As[buf][r][c]), A,
                      ((mBase + r) * Kdim + k0 + c) * 2);
    }
#pragma unroll
    for (int it = 0; it < 2; ++it) {
      int idx = tid + it * 256;
      int r = idx >> 2, c = (idx & 3) * 8;
      async_copy_b128(lds_off(&Bs[buf][r][c]), Wt,
                      ((nBase + r) * Kdim + k0 + c) * 2);
    }
  };

  const int nt = Kdim / BK;
  stage(0, 0);
  for (int kt = 0; kt < nt; ++kt) {
    const int cur = kt & 1;
    if (kt + 1 < nt) {
      stage((kt + 1) * BK, cur ^ 1);
      // async loads complete in order: <=4 outstanding means tile kt landed
      asm volatile("s_wait_asynccnt 4" ::: "memory");
    } else {
      asm volatile("s_wait_asynccnt 0" ::: "memory");
    }
    __syncthreads();

    // A fragment: lanes 0-15 row m, K {0..7,16..23}; lanes 16-31 K {8..15,24..31}
    v16bf af[2], bfr[4];
#pragma unroll
    for (int mi = 0; mi < 2; ++mi) {
      const bf16* ap = &As[cur][wm * 32 + mi * 16 + l16][half * 8];
      af[mi] = concat8(*(const bf16x8*)ap, *(const bf16x8*)(ap + 16));
    }
    // B fragment: lane holds column n; lanes 0-15 K {0..7,16..23}; lanes 16-31 K {8..15,24..31}
#pragma unroll
    for (int ni = 0; ni < 4; ++ni) {
      const bf16* bp = &Bs[cur][wn * 64 + ni * 16 + l16][half * 8];
      bfr[ni] = concat8(*(const bf16x8*)bp, *(const bf16x8*)(bp + 16));
    }
#pragma unroll
    for (int mi = 0; mi < 2; ++mi)
#pragma unroll
      for (int ni = 0; ni < 4; ++ni)
        acc[mi][ni] = wmma_bf16(af[mi], bfr[ni], acc[mi][ni]);
    __syncthreads();
  }

  // Epilogue (compile-time): C layout -> lanes 0-15 rows 0-7, lanes 16-31 rows 8-15
#pragma unroll
  for (int mi = 0; mi < 2; ++mi)
#pragma unroll
    for (int ni = 0; ni < 4; ++ni) {
      const int col = nBase + wn * 64 + ni * 16 + l16;
      const float bv = bias[col];
      if constexpr (EPI == 3) {
        // Transposed store: one lane holds 8 consecutive token rows of column
        // `col` -> a single contiguous 16B store into Vt[B,H,D,S].
        const int row0 = mBase + wm * 32 + mi * 16 + half * 8;
        const int bb = row0 / kS, s0 = row0 % kS;
        const int hh = col >> 6, dd = col & 63;
        bf16x8 vv;
#pragma unroll
        for (int i = 0; i < 8; ++i) vv[i] = (bf16)(acc[mi][ni][i] + bv);
        *(bf16x8*)(Cb + ((size_t)(bb * kH + hh) * kD + dd) * kS + s0) = vv;
      } else {
#pragma unroll
        for (int i = 0; i < 8; ++i) {
          const int row = mBase + wm * 32 + mi * 16 + half * 8 + i;
          const size_t off = (size_t)row * Ndim + col;
          float v = acc[mi][ni][i] + bv;
          if constexpr (EPI == 0) {
            Cb[off] = (bf16)v;
          } else if constexpr (EPI == 1) {
            Cb[off] = (bf16)gelu_exact(v);
          } else {
            Cf[off] = v + resid[off];
          }
        }
      }
    }
}

// ---------------------------------------------------------------------------
// Flash attention (causal). One wave = 16 query rows of one (b,h).
// Q,K are bf16 [B,S,E] (head h at column h*64); V is bf16 [B,H,D,S] so the
// PV B-fragment is K-contiguous in memory. ctx out is bf16 [B,S,E].
// Row sums of exp(P) are accumulated with an extra WMMA against a ones
// fragment (same rescaling as the output accumulator) instead of lane
// shuffles; only the row max needs a 4-stage butterfly.
// ---------------------------------------------------------------------------
__global__ __launch_bounds__(128) void attn_kernel(const bf16* __restrict__ Q,
                                                   const bf16* __restrict__ K,
                                                   const bf16* __restrict__ Vt,
                                                   bf16* __restrict__ Ctx) {
  constexpr int WPB = 4;
  __shared__ __align__(16) bf16 Psh[WPB][16 * 32];   // per-wave P tile
  const int lane = threadIdx.x & 31;
  const int w    = threadIdx.x >> 5;
  const int gw   = blockIdx.x * WPB + w;
  const int qt   = gw % (kS / 16);
  const int bh   = gw / (kS / 16);
  const int h = bh % kH, b = bh / kH;
  const size_t base = (size_t)b * kS * kE + (size_t)h * kD;
  const bf16* Qb = Q + base;
  const bf16* Kb = K + base;
  const bf16* Vtb = Vt + (size_t)(b * kH + h) * kD * kS;   // [D][S]
  bf16* Ob = Ctx + base;
  const int l16  = lane & 15;
  const int half = lane >> 4;

  // Q A-fragments over D=64 (2 k-steps), pre-scaled by 1/sqrt(D)=0.125 (exact)
  v16bf qf[2];
  {
    const bf16* qr = Qb + (size_t)(qt * 16 + l16) * kE;
#pragma unroll
    for (int j = 0; j < 2; ++j) {
      const bf16* p = qr + j * 32 + half * 8;
      bf16x8 lo = *(const bf16x8*)p;
      bf16x8 hi = *(const bf16x8*)(p + 16);
#pragma unroll
      for (int e = 0; e < 8; ++e) {
        qf[j][e]     = (bf16)((float)lo[e] * 0.125f);
        qf[j][8 + e] = (bf16)((float)hi[e] * 0.125f);
      }
    }
  }

  v16bf onesf;
#pragma unroll
  for (int e = 0; e < 16; ++e) onesf[e] = (bf16)1.0f;

  v8f o[4] = {};
  v8f osum = {};                 // row sums of exp(P), C-fragment row layout
  float mrow[8];
#pragma unroll
  for (int i = 0; i < 8; ++i) mrow[i] = -3e30f;

  const int lastChunk = (qt * 16 + 15) >> 5;   // inclusive, 32-key chunks
  for (int c = 0; c <= lastChunk; ++c) {
    // Preload all four K B-fragments (two 16B pieces each), then 4 QK^T WMMAs
    v16bf kf[2][2];
#pragma unroll
    for (int t = 0; t < 2; ++t)
#pragma unroll
      for (int j = 0; j < 2; ++j) {
        const bf16* kp = Kb + (size_t)(c * 32 + t * 16 + l16) * kE + j * 32 + half * 8;
        kf[t][j] = concat8(*(const bf16x8*)kp, *(const bf16x8*)(kp + 16));
      }
    v8f sf[2];
#pragma unroll
    for (int t = 0; t < 2; ++t) {
      v8f accs = {};
      accs = wmma_bf16(qf[0], kf[t][0], accs);
      accs = wmma_bf16(qf[1], kf[t][1], accs);
#pragma unroll
      for (int i = 0; i < 8; ++i) {            // causal mask
        const int qr_ = qt * 16 + half * 8 + i;
        const int kc  = c * 32 + t * 16 + l16;
        if (kc > qr_) accs[i] = -1e9f;
      }
      sf[t] = accs;
    }
    // Online softmax across the 32-key chunk; write exp(P) to LDS as bf16
#pragma unroll
    for (int i = 0; i < 8; ++i) {
      float vmax = fmaxf(sf[0][i], sf[1][i]);
#pragma unroll
      for (int off = 1; off < 16; off <<= 1) vmax = fmaxf(vmax, __shfl_xor(vmax, off, 16));
      const float mnew  = fmaxf(mrow[i], vmax);
      const float scale = __expf(mrow[i] - mnew);
      const float p0 = __expf(sf[0][i] - mnew);
      const float p1 = __expf(sf[1][i] - mnew);
      mrow[i] = mnew;
      osum[i] *= scale;
#pragma unroll
      for (int g = 0; g < 4; ++g) o[g][i] *= scale;
      const int r = half * 8 + i;
      Psh[w][r * 32 + l16]      = (bf16)p0;
      Psh[w][r * 32 + 16 + l16] = (bf16)p1;
    }
    // P stores must land before the layout-changing re-read (wave-local order)
    asm volatile("s_wait_dscnt 0" ::: "memory");
    // P as A-fragment (two 16B LDS vector loads)
    v16bf pf;
    {
      const bf16* pp = &Psh[w][l16 * 32 + half * 8];
      pf = concat8(*(const bf16x8*)pp, *(const bf16x8*)(pp + 16));
    }
    // Row sums via matrix op: osum += P @ ones (replaces 32 lane shuffles)
    osum = wmma_bf16(pf, onesf, osum);
    // P @ V: V is [D][S] so each B-fragment is keys {8h..8h+7, 16+8h..16+8h+7} (16B loads)
#pragma unroll
    for (int g = 0; g < 4; ++g) {
      const bf16* vp = Vtb + (size_t)(g * 16 + l16) * kS + c * 32 + half * 8;
      v16bf vf = concat8(*(const bf16x8*)vp, *(const bf16x8*)(vp + 16));
      o[g] = wmma_bf16(pf, vf, o[g]);
    }
  }
  // Normalize and store ctx (bf16, [B,S,E] layout)
#pragma unroll
  for (int i = 0; i < 8; ++i) {
    const float inv = 1.f / osum[i];
    const int qr_ = qt * 16 + half * 8 + i;
    bf16* orow = Ob + (size_t)qr_ * kE;
#pragma unroll
    for (int g = 0; g < 4; ++g) orow[g * 16 + l16] = (bf16)(o[g][i] * inv);
  }
}

// ---------------------------------------------------------------------------
// Launcher
// ---------------------------------------------------------------------------
extern "C" void kernel_launch(void* const* d_in, const int* in_sizes, int n_in,
                              void* d_out, int out_size, void* d_ws, size_t ws_size,
                              hipStream_t stream) {
  (void)in_sizes; (void)n_in; (void)out_size; (void)ws_size;
  const float* x    = (const float*)d_in[0];
  // d_in[1] = mask (unused; causality applied analytically)
  const float* Wq   = (const float*)d_in[2];
  const float* bq   = (const float*)d_in[3];
  const float* Wk   = (const float*)d_in[4];
  const float* bk   = (const float*)d_in[5];
  const float* Wv   = (const float*)d_in[6];
  const float* bv   = (const float*)d_in[7];
  const float* Wo   = (const float*)d_in[8];
  const float* bo   = (const float*)d_in[9];
  const float* W1   = (const float*)d_in[10];
  const float* b1   = (const float*)d_in[11];
  const float* W2   = (const float*)d_in[12];
  const float* b2   = (const float*)d_in[13];
  const float* ln1g = (const float*)d_in[14];
  const float* ln1b = (const float*)d_in[15];
  const float* ln2g = (const float*)d_in[16];
  const float* ln2b = (const float*)d_in[17];

  char* p = (char*)d_ws;
  auto carve = [&p](size_t bytes) -> char* {
    char* r = p; p += (bytes + 255) & ~(size_t)255; return r;
  };
  bf16* Wq_t  = (bf16*)carve((size_t)kE * kE * sizeof(bf16));   // [N][K]
  bf16* Wk_t  = (bf16*)carve((size_t)kE * kE * sizeof(bf16));
  bf16* Wv_t  = (bf16*)carve((size_t)kE * kE * sizeof(bf16));
  bf16* Wo_t  = (bf16*)carve((size_t)kE * kE * sizeof(bf16));
  bf16* W1_t  = (bf16*)carve((size_t)kE * kF * sizeof(bf16));
  bf16* W2_t  = (bf16*)carve((size_t)kF * kE * sizeof(bf16));
  bf16* n_b   = (bf16*)carve((size_t)kM * kE * sizeof(bf16));
  bf16* Q_b   = (bf16*)carve((size_t)kM * kE * sizeof(bf16));
  bf16* K_b   = (bf16*)carve((size_t)kM * kE * sizeof(bf16));
  bf16* Vt_b  = (bf16*)carve((size_t)kM * kE * sizeof(bf16));   // [B,H,D,S]
  bf16* ctx_b = (bf16*)carve((size_t)kM * kE * sizeof(bf16));
  float* y_f  = (float*)carve((size_t)kM * kE * sizeof(float));
  float* x2_f = (float*)carve((size_t)kM * kE * sizeof(float));
  bf16* x2_b  = (bf16*)carve((size_t)kM * kE * sizeof(bf16));
  bf16* h1_b  = (bf16*)carve((size_t)kM * kF * sizeof(bf16));

  auto cvt_tr = [&](const float* src, bf16* dst, int K, int N) {
    size_t n = (size_t)K * N;
    cvt_tr_bf16_kernel<<<(unsigned)((n + 255) / 256), 256, 0, stream>>>(src, dst, K, N);
  };
  cvt_tr(Wq, Wq_t, kE, kE);
  cvt_tr(Wk, Wk_t, kE, kE);
  cvt_tr(Wv, Wv_t, kE, kE);
  cvt_tr(Wo, Wo_t, kE, kE);
  cvt_tr(W1, W1_t, kE, kF);
  cvt_tr(W2, W2_t, kF, kE);

  // ln1 -> n (bf16)
  ln_kernel<kE><<<kM, 256, 0, stream>>>(x, ln1g, ln1b, nullptr, n_b);

  // QKV projections (bf16 WMMA, bias fused). V lands transposed [B,H,D,S].
  dim3 gE(kE / 128, kM / 128);
  gemm_bf16_kernel<0><<<gE, 256, 0, stream>>>(n_b, Wq_t, bq, nullptr, nullptr, Q_b,
                                              kM, kE, kE);
  gemm_bf16_kernel<0><<<gE, 256, 0, stream>>>(n_b, Wk_t, bk, nullptr, nullptr, K_b,
                                              kM, kE, kE);
  gemm_bf16_kernel<3><<<gE, 256, 0, stream>>>(n_b, Wv_t, bv, nullptr, nullptr, Vt_b,
                                              kM, kE, kE);

  // Causal flash attention -> ctx (bf16)
  attn_kernel<<<(kB * kH * (kS / 16)) / 4, 128, 0, stream>>>(Q_b, K_b, Vt_b, ctx_b);

  // Out projection + residual: y = x + ctx @ Wo + bo (fp32)
  gemm_bf16_kernel<2><<<gE, 256, 0, stream>>>(ctx_b, Wo_t, bo, x, y_f, nullptr,
                                              kM, kE, kE);

  // ln2 -> x2 (fp32 for residual, bf16 for GEMM)
  ln_kernel<kE><<<kM, 256, 0, stream>>>(y_f, ln2g, ln2b, x2_f, x2_b);

  // FFN1: h1 = gelu(x2 @ W1 + b1) (bf16)
  dim3 gF(kF / 128, kM / 128);
  gemm_bf16_kernel<1><<<gF, 256, 0, stream>>>(x2_b, W1_t, b1, nullptr, nullptr, h1_b,
                                              kM, kF, kE);

  // FFN2 + residual: out = x2 + h1 @ W2 + b2 (fp32 -> d_out)
  gemm_bf16_kernel<2><<<gE, 256, 0, stream>>>(h1_b, W2_t, b2, x2_f, (float*)d_out,
                                              nullptr, kM, kE, kF);
}
